// CausalSelfAttention_58660663329038
// MI455X (gfx1250) — compile-verified, matrix operand order checked by reading, not run on hardware
//
#include <hip/hip_runtime.h>
#include <hip/hip_bf16.h>

typedef __attribute__((ext_vector_type(16))) __bf16 v16bf;
typedef __attribute__((ext_vector_type(8)))  float  v8f;
typedef unsigned short u16;

#define NB 4
#define NS 2048
#define ND 1024
#define NH 16
#define HD 64

static __device__ __forceinline__ v8f wmma_bf16(v16bf a, v16bf b, v8f c) {
  return __builtin_amdgcn_wmma_f32_16x16x32_bf16(false, a, false, b, (short)0, c, false, false);
}

static __device__ __forceinline__ u16 f32_to_bf16_bits(float f) {
  union { __bf16 h; u16 u; } cv;
  cv.h = (__bf16)f;
  return cv.u;
}

// WMMA bf16 fragment for NT GEMMs (both operands contiguous along K).
// Per-lane K chunks: {kb+8*half .. +7} and {kb+16+8*half .. +7} -> 2x b128.
static __device__ __forceinline__ v16bf load_frag_bf16(const u16* __restrict__ row,
                                                       int kb, int half) {
  const u16* p = row + kb + 8 * half;
  union { v16bf v; uint4 q[2]; } f;
  f.q[0] = *(const uint4*)(p);
  f.q[1] = *(const uint4*)(p + 16);
  return f.v;
}

// ---------------------------------------------------------------------------
// One-shot fp32 -> bf16 conversion (8 elements/thread, b128 in / b128 out).
// ---------------------------------------------------------------------------
__global__ __launch_bounds__(256) void cvt_bf16_kernel(const float* __restrict__ src,
                                                       u16* __restrict__ dst, int n8) {
  const int i = blockIdx.x * 256 + threadIdx.x;
  if (i >= n8) return;
  const float4* s = (const float4*)src + (size_t)i * 2;
  float4 f0 = s[0], f1 = s[1];
  union { u16 h[8]; uint4 q; } u;
  u.h[0] = f32_to_bf16_bits(f0.x); u.h[1] = f32_to_bf16_bits(f0.y);
  u.h[2] = f32_to_bf16_bits(f0.z); u.h[3] = f32_to_bf16_bits(f0.w);
  u.h[4] = f32_to_bf16_bits(f1.x); u.h[5] = f32_to_bf16_bits(f1.y);
  u.h[6] = f32_to_bf16_bits(f1.z); u.h[7] = f32_to_bf16_bits(f1.w);
  *((uint4*)dst + i) = u.q;
}

// ---------------------------------------------------------------------------
// GEMM helpers: one k-chunk = A fragment (shared) + 8 B fragments (N tiles).
// ---------------------------------------------------------------------------
static __device__ __forceinline__ void load_gemm_chunk(
    const u16* __restrict__ xrow, const u16* __restrict__ Wbase,
    int kb, int half, v16bf& a, v16bf b[8]) {
  a = load_frag_bf16(xrow, kb, half);
  #pragma unroll
  for (int t = 0; t < 8; ++t)
    b[t] = load_frag_bf16(Wbase + (size_t)t * 16 * ND, kb, half);
}

static __device__ __forceinline__ void wmma_all(const v16bf& a, const v16bf b[8],
                                                v8f acc[8]) {
  #pragma unroll
  for (int t = 0; t < 8; ++t) acc[t] = wmma_bf16(a, b[t], acc[t]);
}

// ---------------------------------------------------------------------------
// Kernel A: fused QKV projection. out = x @ W^T + b, all-bf16 NT WMMA GEMM.
// One wave computes 16(M) x 128(N); double-buffered fragment sets pipeline
// global loads against WMMAs. Q,K stored [B,H,S,HD] bf16; V transposed.
// ---------------------------------------------------------------------------
__global__ __launch_bounds__(256) void qkv_proj_kernel(
    const u16* __restrict__ xb,
    const u16* __restrict__ Wqb, const float* __restrict__ bq,
    const u16* __restrict__ Wkb, const float* __restrict__ bk,
    const u16* __restrict__ Wvb, const float* __restrict__ bv,
    u16* __restrict__ Qb, u16* __restrict__ Kb, u16* __restrict__ Vt)
{
  const int lane = threadIdx.x & 31;
  const int wave = threadIdx.x >> 5;
  const int idx  = lane & 15;
  const int half = lane >> 4;

  int tile = blockIdx.x * 8 + wave;          // 3 * 512 * 8 wave tiles
  const int which = tile / (512 * 8);        // 0=Q 1=K 2=V
  tile -= which * (512 * 8);
  const int mt = tile >> 3;                  // 512 row tiles of 16
  const int nt = tile & 7;                   // 8 col tiles of 128

  const u16*   __restrict__ W    = (which == 0) ? Wqb : (which == 1) ? Wkb : Wvb;
  const float* __restrict__ bias = (which == 0) ? bq  : (which == 1) ? bk  : bv;

  const u16* xrow  = xb + (size_t)(mt * 16 + idx) * ND;
  const u16* Wbase = W + (size_t)(nt * 128 + idx) * ND;

  v8f acc[8] = {};
  v16bf aA, bA[8], aB, bB[8];
  load_gemm_chunk(xrow, Wbase, 0, half, aA, bA);
  for (int kb = 0; kb < ND; kb += 64) {
    load_gemm_chunk(xrow, Wbase, kb + 32, half, aB, bB);
    wmma_all(aA, bA, acc);
    if (kb + 64 < ND) load_gemm_chunk(xrow, Wbase, kb + 64, half, aA, bA);
    wmma_all(aB, bB, acc);
  }

  #pragma unroll
  for (int t = 0; t < 8; ++t) {
    const int n  = nt * 128 + t * 16 + idx;
    const float bn = bias[n];
    const int h = n >> 6, d = n & 63;
    #pragma unroll
    for (int r = 0; r < 8; ++r) {
      const int mrow = mt * 16 + r + 8 * half;   // C/D layout: M = r + 8*half
      const int bb = mrow >> 11;                 // / NS
      const int s  = mrow & (NS - 1);
      const u16 val = f32_to_bf16_bits(acc[t][r] + bn);
      if (which == 0)
        Qb[(((size_t)(bb * NH + h) * NS) + s) * HD + d] = val;
      else if (which == 1)
        Kb[(((size_t)(bb * NH + h) * NS) + s) * HD + d] = val;
      else
        Vt[(((size_t)(bb * NH + h) * HD) + d) * NS + s] = val;
    }
  }
}

// ---------------------------------------------------------------------------
// Kernel B: causal flash attention. One wave owns 16 query rows; workgroup of
// 8 waves covers 128 rows of one (b,h). Key blocks of 32; online softmax with
// width-16 shuffles; P converted C-layout -> A-layout through per-wave LDS.
// Pipelined: V frags load at top of iter (consumed after softmax); next K
// frags load right after the score WMMAs release the current ones.
// ---------------------------------------------------------------------------
static __device__ __forceinline__ void load_k_frags(const u16* __restrict__ Kh,
                                                    int kk, int idx, int half,
                                                    v16bf kf[4]) {
  kf[0] = load_frag_bf16(Kh + (size_t)(kk      + idx) * HD, 0,  half);
  kf[1] = load_frag_bf16(Kh + (size_t)(kk      + idx) * HD, 32, half);
  kf[2] = load_frag_bf16(Kh + (size_t)(kk + 16 + idx) * HD, 0,  half);
  kf[3] = load_frag_bf16(Kh + (size_t)(kk + 16 + idx) * HD, 32, half);
}

__global__ __launch_bounds__(256) void attn_kernel(
    const u16* __restrict__ Qb, const u16* __restrict__ Kb,
    const u16* __restrict__ Vt, u16* __restrict__ O)
{
  __shared__ alignas(16) u16 pbuf[8][16][40];   // stride 80B: 16B-aligned chunks
  const int lane = threadIdx.x & 31;
  const int wave = threadIdx.x >> 5;
  const int idx  = lane & 15;
  const int half = lane >> 4;

  const int bh = blockIdx.x >> 4;                         // (b*NH + h)
  const int q0 = ((blockIdx.x & 15) * 8 + wave) * 16;

  const u16* Qh = Qb + (size_t)bh * NS * HD;
  const u16* Kh = Kb + (size_t)bh * NS * HD;
  const u16* Vh = Vt + (size_t)bh * HD * NS;

  const v16bf aq0 = load_frag_bf16(Qh + (size_t)(q0 + idx) * HD, 0,  half);
  const v16bf aq1 = load_frag_bf16(Qh + (size_t)(q0 + idx) * HD, 32, half);

  v8f o0 = {}, o1 = {}, o2 = {}, o3 = {};
  float mrow[8], lrow[8];
  #pragma unroll
  for (int r = 0; r < 8; ++r) { mrow[r] = -__builtin_inff(); lrow[r] = 0.f; }

  const float scale = 2.8284271247461903f;   // HD**0.25 (reference quirk)
  const int nblocks = ((q0 + 15) >> 5) + 1;  // causal: cover cols 0..q0+15

  v16bf kc[4];
  load_k_frags(Kh, 0, idx, half, kc);

  for (int jb = 0; jb < nblocks; ++jb) {
    const int kk = jb * 32;
    // ---- V frags for this block: issued early, consumed after softmax ----
    v16bf vf0 = load_frag_bf16(Vh + (size_t)( 0 + idx) * NS, kk, half);
    v16bf vf1 = load_frag_bf16(Vh + (size_t)(16 + idx) * NS, kk, half);
    v16bf vf2 = load_frag_bf16(Vh + (size_t)(32 + idx) * NS, kk, half);
    v16bf vf3 = load_frag_bf16(Vh + (size_t)(48 + idx) * NS, kk, half);

    // ---- scores: 16x32 tile = two 16x16 WMMA outputs, K=64 chained ----
    v8f s0 = {}, s1 = {};
    s0 = wmma_bf16(aq0, kc[0], s0);
    s0 = wmma_bf16(aq1, kc[1], s0);
    s1 = wmma_bf16(aq0, kc[2], s1);
    s1 = wmma_bf16(aq1, kc[3], s1);

    // ---- pipeline: next block's K frags + far prefetch ----
    if (jb + 1 < nblocks) load_k_frags(Kh, kk + 32, idx, half, kc);
    __builtin_prefetch(Kh + (size_t)(kk + 64 + idx) * HD, 0, 3);
    __builtin_prefetch(Vh + (size_t)idx * NS + kk + 64, 0, 3);

    // ---- scale + causal mask + online softmax update ----
    #pragma unroll
    for (int r = 0; r < 8; ++r) {
      const int q = q0 + r + 8 * half;
      float v0 = (kk      + idx <= q) ? s0[r] * scale : -__builtin_inff();
      float v1 = (kk + 16 + idx <= q) ? s1[r] * scale : -__builtin_inff();
      float rm = fmaxf(v0, v1);
      rm = fmaxf(rm, __shfl_xor(rm, 1, 16));
      rm = fmaxf(rm, __shfl_xor(rm, 2, 16));
      rm = fmaxf(rm, __shfl_xor(rm, 4, 16));
      rm = fmaxf(rm, __shfl_xor(rm, 8, 16));
      const float mn    = fmaxf(mrow[r], rm);
      const float alpha = __expf(mrow[r] - mn);
      const float p0 = __expf(v0 - mn);
      const float p1 = __expf(v1 - mn);
      float rs = p0 + p1;
      rs += __shfl_xor(rs, 1, 16);
      rs += __shfl_xor(rs, 2, 16);
      rs += __shfl_xor(rs, 4, 16);
      rs += __shfl_xor(rs, 8, 16);
      lrow[r] = lrow[r] * alpha + rs;
      mrow[r] = mn;
      o0[r] *= alpha; o1[r] *= alpha; o2[r] *= alpha; o3[r] *= alpha;
      // stash P (bf16) in C/D layout: element (M=r+8*half, N)
      const int prow = r + 8 * half;
      pbuf[wave][prow][idx]      = f32_to_bf16_bits(p0);
      pbuf[wave][prow][16 + idx] = f32_to_bf16_bits(p1);
    }
    asm volatile("s_wait_dscnt 0x0" ::: "memory");   // same-wave LDS RAW
    // ---- reload P in A-operand layout: row=idx, K chunks by half ----
    union { v16bf v; uint4 q[2]; } pf;
    pf.q[0] = *(const uint4*)&pbuf[wave][idx][8 * half];
    pf.q[1] = *(const uint4*)&pbuf[wave][idx][16 + 8 * half];
    // ---- O += P @ V  (NT against transposed V rows, contiguous in j) ----
    o0 = wmma_bf16(pf.v, vf0, o0);
    o1 = wmma_bf16(pf.v, vf1, o1);
    o2 = wmma_bf16(pf.v, vf2, o2);
    o3 = wmma_bf16(pf.v, vf3, o3);
  }

  const int bb = bh >> 4, h = bh & 15;
  #pragma unroll
  for (int r = 0; r < 8; ++r) {
    const float inv = 1.f / lrow[r];
    const int q = q0 + r + 8 * half;
    u16* orow = O + ((size_t)bb * NS + q) * ND + h * HD;
    orow[ 0 + idx] = f32_to_bf16_bits(o0[r] * inv);
    orow[16 + idx] = f32_to_bf16_bits(o1[r] * inv);
    orow[32 + idx] = f32_to_bf16_bits(o2[r] * inv);
    orow[48 + idx] = f32_to_bf16_bits(o3[r] * inv);
  }
}

// ---------------------------------------------------------------------------
// Kernel C: output projection. d_out = O @ Wo^T + bo (bf16 in, fp32 out).
// One wave computes a 16(M) x 128(N) tile, double-buffered like kernel A.
// ---------------------------------------------------------------------------
__global__ __launch_bounds__(256) void out_proj_kernel(
    const u16* __restrict__ O, const u16* __restrict__ Wob,
    const float* __restrict__ bo, float* __restrict__ out)
{
  const int lane = threadIdx.x & 31;
  const int wave = threadIdx.x >> 5;
  const int idx  = lane & 15;
  const int half = lane >> 4;

  const int tile = blockIdx.x * 8 + wave;    // 512 * 8 wave tiles
  const int mt = tile >> 3;
  const int nt = tile & 7;

  const u16* orow  = O + (size_t)(mt * 16 + idx) * ND;
  const u16* Wbase = Wob + (size_t)(nt * 128 + idx) * ND;

  v8f acc[8] = {};
  v16bf aA, bA[8], aB, bB[8];
  load_gemm_chunk(orow, Wbase, 0, half, aA, bA);
  for (int kb = 0; kb < ND; kb += 64) {
    load_gemm_chunk(orow, Wbase, kb + 32, half, aB, bB);
    wmma_all(aA, bA, acc);
    if (kb + 64 < ND) load_gemm_chunk(orow, Wbase, kb + 64, half, aA, bA);
    wmma_all(aB, bB, acc);
  }

  #pragma unroll
  for (int t = 0; t < 8; ++t) {
    const int n = nt * 128 + t * 16 + idx;
    const float bn = bo[n];
    #pragma unroll
    for (int r = 0; r < 8; ++r) {
      const int mrow = mt * 16 + r + 8 * half;
      out[(size_t)mrow * ND + n] = acc[t][r] + bn;
    }
  }
}

// ---------------------------------------------------------------------------
extern "C" void kernel_launch(void* const* d_in, const int* in_sizes, int n_in,
                              void* d_out, int out_size, void* d_ws, size_t ws_size,
                              hipStream_t stream) {
  const float* x  = (const float*)d_in[0];
  const float* Wq = (const float*)d_in[1];
  const float* bq = (const float*)d_in[2];
  const float* Wk = (const float*)d_in[3];
  const float* bk = (const float*)d_in[4];
  const float* Wv = (const float*)d_in[5];
  const float* bv = (const float*)d_in[6];
  const float* Wo = (const float*)d_in[7];
  const float* bo = (const float*)d_in[8];
  float* out = (float*)d_out;

  char* ws = (char*)d_ws;
  const size_t x_elems = (size_t)NB * NS * ND;      // 8 Mi
  const size_t w_elems = (size_t)ND * ND;           // 1 Mi
  const size_t qkv_elems = (size_t)NB * NH * NS * HD;

  u16* xb  = (u16*)(ws);                                   // 16 MB
  u16* Wqb = (u16*)(ws + 2 * x_elems);                     //  2 MB
  u16* Wkb = Wqb + w_elems;
  u16* Wvb = Wkb + w_elems;
  u16* Wob = Wvb + w_elems;
  u16* Qb  = Wob + w_elems;                                // 16 MB
  u16* Kb  = Qb + qkv_elems;
  u16* Vt  = Kb + qkv_elems;
  u16* Ob  = Vt + qkv_elems;                               // 16 MB

  // one-shot bf16 conversion of activations and weights
  cvt_bf16_kernel<<<(int)(x_elems / 8 / 256), 256, 0, stream>>>(x, xb, (int)(x_elems / 8));
  cvt_bf16_kernel<<<(int)(w_elems / 8 / 256), 256, 0, stream>>>(Wq, Wqb, (int)(w_elems / 8));
  cvt_bf16_kernel<<<(int)(w_elems / 8 / 256), 256, 0, stream>>>(Wk, Wkb, (int)(w_elems / 8));
  cvt_bf16_kernel<<<(int)(w_elems / 8 / 256), 256, 0, stream>>>(Wv, Wvb, (int)(w_elems / 8));
  cvt_bf16_kernel<<<(int)(w_elems / 8 / 256), 256, 0, stream>>>(Wo, Wob, (int)(w_elems / 8));

  qkv_proj_kernel<<<3 * 512 * 8 / 8, 256, 0, stream>>>(
      xb, Wqb, bq, Wkb, bk, Wvb, bv, Qb, Kb, Vt);
  attn_kernel<<<NB * NH * (NS / 128), 256, 0, stream>>>(Qb, Kb, Vt, Ob);
  out_proj_kernel<<<512 * 8 / 8, 256, 0, stream>>>(Ob, Wob, bo, out);
}
